// BasicBlock_1047972020336
// MI455X (gfx1250) — 8-bit matrix operand order repaired and checked by reading, compiled, not run on hardware
//
#include <hip/hip_runtime.h>
#include <hip/hip_bf16.h>
#include <stdint.h>


typedef int v8i __attribute__((ext_vector_type(8)));
typedef int gv4i __attribute__((vector_size(16)));
typedef __attribute__((address_space(1))) gv4i* gv4i_gptr;
typedef __attribute__((address_space(3))) gv4i* gv4i_sptr;

#define NB  32
#define CCH 256
#define HH  56
#define WW  56
#define HP  58
#define WP  58
#define EPSV 1e-5f

#if defined(__has_builtin)
#if __has_builtin(__builtin_amdgcn_global_load_async_to_lds_b128) && \
    __has_builtin(__builtin_amdgcn_s_wait_asynccnt)
#define HAVE_ASYNC_LDS 1
#endif
#endif

__device__ __forceinline__ int8_t sgn8(float v) {
    return (int8_t)((v > 0.f) - (v < 0.f));
}

__global__ __launch_bounds__(256) void k_consts(
    const float* __restrict__ w3, const float* __restrict__ w1,
    const float* __restrict__ g1, const float* __restrict__ be1,
    const float* __restrict__ m1, const float* __restrict__ v1,
    const float* __restrict__ g2, const float* __restrict__ be2,
    const float* __restrict__ m2, const float* __restrict__ v2,
    float* __restrict__ sc1, float* __restrict__ bs1,
    float* __restrict__ sc2, float* __restrict__ bs2)
{
    const int o = threadIdx.x;
    float s3 = 0.f;
    const float* p3 = w3 + (size_t)o * 2304;
    for (int i = 0; i < 2304; ++i) s3 += fabsf(p3[i]);
    s3 *= (1.f / 2304.f);
    float s1 = 0.f;
    const float* p1 = w1 + (size_t)o * 256;
    for (int i = 0; i < 256; ++i) s1 += fabsf(p1[i]);
    s1 *= (1.f / 256.f);
    const float i1 = g1[o] * rsqrtf(v1[o] + EPSV);
    const float i2 = g2[o] * rsqrtf(v2[o] + EPSV);
    sc1[o] = s3 * i1; bs1[o] = be1[o] - m1[o] * i1;
    sc2[o] = s1 * i2; bs2[o] = be2[o] - m2[o] * i2;
}

__global__ __launch_bounds__(256) void k_pack3(const float* __restrict__ w3,
                                               uint32_t* __restrict__ B3)
{
    const int gid  = blockIdx.x * 256 + threadIdx.x;
    const int v    = gid & 7;
    const int lane = (gid >> 3) & 31;
    const int ntkc = gid >> 8;
    const int nt   = ntkc & 15, kc = ntkc >> 4;
    const int N    = nt * 16 + (lane & 15);
    const int kb   = kc * 64 + ((v >> 1) << 4) + ((lane >> 4) << 3) + ((v & 1) << 2);
    uint32_t dw = 0;
#pragma unroll
    for (int b = 0; b < 4; ++b) {
        const int K = kb + b;
        const int tap = K >> 8, c = K & 255;
        const float wv = w3[((size_t)N * 256 + c) * 9 + tap];
        dw |= (uint32_t)(uint8_t)sgn8(wv) << (8 * b);
    }
    B3[gid] = dw;
}

__global__ __launch_bounds__(256) void k_pack1(const float* __restrict__ w1,
                                               uint32_t* __restrict__ B1)
{
    const int gid  = blockIdx.x * 256 + threadIdx.x;
    const int v    = gid & 7;
    const int lane = (gid >> 3) & 31;
    const int ntkc = gid >> 8;
    const int nt   = ntkc & 15, kc = ntkc >> 4;
    const int N    = nt * 16 + (lane & 15);
    const int kb   = kc * 64 + ((v >> 1) << 4) + ((lane >> 4) << 3) + ((v & 1) << 2);
    uint32_t dw = 0;
#pragma unroll
    for (int b = 0; b < 4; ++b) {
        const int c = kb + b;
        const float wv = w1[(size_t)N * 256 + c];
        dw |= (uint32_t)(uint8_t)sgn8(wv) << (8 * b);
    }
    B1[gid] = dw;
}

__global__ __launch_bounds__(256) void k_binact(const float* __restrict__ x,
                                                const float* __restrict__ b11,
                                                uint32_t* __restrict__ a1bin)
{
    const int gid = blockIdx.x * 256 + threadIdx.x;
    const int c4 = gid & 63;
    int t = gid >> 6;
    const int wp = t % WP; t /= WP;
    const int hp = t % HP; const int n = t / HP;
    uint32_t dw = 0;
    if (hp >= 1 && hp <= HH && wp >= 1 && wp <= WW) {
        const int h = hp - 1, w = wp - 1;
#pragma unroll
        for (int k = 0; k < 4; ++k) {
            const int c = c4 * 4 + k;
            const float xv = x[(((size_t)n * CCH + c) * HH + h) * WW + w] + b11[c];
            dw |= (uint32_t)(uint8_t)sgn8(xv) << (8 * k);
        }
    }
    a1bin[gid] = dw;
}

__device__ __forceinline__ v8i load_a_frag(const uint8_t* __restrict__ rp) {
    const uint2 q0 = *(const uint2*)(rp);
    const uint2 q1 = *(const uint2*)(rp + 16);
    const uint2 q2 = *(const uint2*)(rp + 32);
    const uint2 q3 = *(const uint2*)(rp + 48);
    v8i a = { (int)q0.x, (int)q0.y, (int)q1.x, (int)q1.y,
              (int)q2.x, (int)q2.y, (int)q3.x, (int)q3.y };
    return a;
}

__device__ __forceinline__ v8i load_b_frag(const int* __restrict__ p) {
    const int4 u0 = *(const int4*)(p);
    const int4 u1 = *(const int4*)(p + 4);
    v8i b = { u0.x, u0.y, u0.z, u0.w, u1.x, u1.y, u1.z, u1.w };
    return b;
}

#define WMMA_IU8(A, B, C) \
    __builtin_amdgcn_wmma_i32_16x16x64_iu8(true, (A), true, (B), (C), false, false)

__global__ __launch_bounds__(256) void k_conv1(
    const uint8_t* __restrict__ a1bin, const int* __restrict__ B3,
    const float* __restrict__ x,
    const float* __restrict__ sc1, const float* __restrict__ bs1,
    const float* __restrict__ b12, const float* __restrict__ b13,
    const float* __restrict__ a1,  const float* __restrict__ b21,
    float* __restrict__ out1f, int8_t* __restrict__ out1bin)
{
    __shared__ uint8_t patch[6 * 10 * 256];

    const int tid   = threadIdx.x;
    const int lane  = tid & 31, wave = tid >> 5;
    const int waveN = wave & 3, waveM = wave >> 2;
    const int n  = blockIdx.z;
    const int h0 = blockIdx.y * 4;
    const int w0 = blockIdx.x * 8;

    {
        const size_t gq = ((size_t)n * HP + h0) * WP + w0;
        for (int i = tid; i < 960; i += 256) {
            const int s = i >> 4, o = (i & 15) * 16;
            const int dy = s / 10, dx = s - dy * 10;
            const uint8_t* src = a1bin + (gq + (size_t)dy * WP + dx) * 256 + o;
            uint8_t*       dst = patch + s * 256 + o;
#if HAVE_ASYNC_LDS
            __builtin_amdgcn_global_load_async_to_lds_b128(
                (gv4i_gptr)src, (gv4i_sptr)dst, 0, 0);
#else
            *(int4*)dst = *(const int4*)src;
#endif
        }
#if HAVE_ASYNC_LDS
        __builtin_amdgcn_s_wait_asynccnt(0);
#endif
    }
    __syncthreads();

    const int g  = lane >> 4;
    const int m  = lane & 15;
    const int py = m >> 3, px = m & 7;
    const int arow = waveM * 2 + py;

    v8i acc0 = {}, acc1 = {}, acc2 = {}, acc3 = {};

    for (int kc = 0; kc < 36; ++kc) {
        const int tap = kc >> 2, cc = kc & 3;
        const int kh = tap / 3, kw = tap - kh * 3;
        const uint8_t* rp = patch + ((arow + kh) * 10 + (px + kw)) * 256
                                  + cc * 64 + g * 8;
        const v8i a = load_a_frag(rp);

        const int* bp = B3 + ((size_t)(kc * 16 + waveN * 4) * 32 + lane) * 8;
        __builtin_prefetch(bp + 4096, 0, 3);
        const v8i b0 = load_b_frag(bp);
        const v8i b1 = load_b_frag(bp + 256);
        const v8i b2 = load_b_frag(bp + 512);
        const v8i b3 = load_b_frag(bp + 768);

        acc0 = WMMA_IU8(a, b0, acc0);
        acc1 = WMMA_IU8(a, b1, acc1);
        acc2 = WMMA_IU8(a, b2, acc2);
        acc3 = WMMA_IU8(a, b3, acc3);
    }

    const int hrow = h0 + waveM * 2 + g;
    const int ocl  = lane & 15;
    const v8i accs[4] = {acc0, acc1, acc2, acc3};
#pragma unroll
    for (int nt = 0; nt < 4; ++nt) {
        const int oc = (waveN * 4 + nt) * 16 + ocl;
        const float sc = sc1[oc], bi = bs1[oc];
        const float c12 = b12[oc], c13 = b13[oc], al = a1[oc], c21 = b21[oc];
        const size_t xb = (((size_t)n * CCH + oc) * HH + hrow) * WW + w0;
        const size_t bb = (((size_t)n * HH + hrow) * WW + w0) * 256 + oc;
#pragma unroll
        for (int j = 0; j < 8; ++j) {
            const float f   = (float)accs[nt][j] * sc + bi;
            const float o1  = x[xb + j] + f;
            const float t   = o1 + c12;
            const float o1v = (t > 0.f ? t : al * t) + c13;
            out1f[xb + j] = o1v;
            out1bin[bb + (size_t)j * 256] = sgn8(o1v + c21);
        }
    }
}

__global__ __launch_bounds__(256) void k_conv2(
    const uint8_t* __restrict__ obin, const int* __restrict__ B1,
    const float* __restrict__ out1f,
    const float* __restrict__ sc2, const float* __restrict__ bs2,
    const float* __restrict__ b22, const float* __restrict__ b23,
    const float* __restrict__ a2,
    float* __restrict__ out)
{
    const int tid   = threadIdx.x;
    const int lane  = tid & 31, wave = tid >> 5;
    const int waveN = wave & 3, waveM = wave >> 2;
    const int n  = blockIdx.z;
    const int h0 = blockIdx.y * 4;
    const int w0 = blockIdx.x * 8;

    const int g  = lane >> 4;
    const int m  = lane & 15;
    const int py = m >> 3, px = m & 7;
    const int h  = h0 + waveM * 2 + py;
    const int w  = w0 + px;

    const uint8_t* rowbase = obin + (((size_t)n * HH + h) * WW + w) * 256 + g * 8;

    v8i acc0 = {}, acc1 = {}, acc2 = {}, acc3 = {};

#pragma unroll
    for (int kc = 0; kc < 4; ++kc) {
        const v8i a = load_a_frag(rowbase + kc * 64);
        const int* bp = B1 + ((size_t)(kc * 16 + waveN * 4) * 32 + lane) * 8;
        const v8i b0 = load_b_frag(bp);
        const v8i b1 = load_b_frag(bp + 256);
        const v8i b2 = load_b_frag(bp + 512);
        const v8i b3 = load_b_frag(bp + 768);
        acc0 = WMMA_IU8(a, b0, acc0);
        acc1 = WMMA_IU8(a, b1, acc1);
        acc2 = WMMA_IU8(a, b2, acc2);
        acc3 = WMMA_IU8(a, b3, acc3);
    }

    const int hrow = h0 + waveM * 2 + g;
    const int ocl  = lane & 15;
    const v8i accs[4] = {acc0, acc1, acc2, acc3};
#pragma unroll
    for (int nt = 0; nt < 4; ++nt) {
        const int oc = (waveN * 4 + nt) * 16 + ocl;
        const float sc = sc2[oc], bi = bs2[oc];
        const float c22 = b22[oc], c23 = b23[oc], al = a2[oc];
        const size_t xb = (((size_t)n * CCH + oc) * HH + hrow) * WW + w0;
#pragma unroll
        for (int j = 0; j < 8; ++j) {
            const float f  = (float)accs[nt][j] * sc + bi;
            const float o2 = f + out1f[xb + j];
            const float t  = o2 + c22;
            out[xb + j] = (t > 0.f ? t : al * t) + c23;
        }
    }
}

extern "C" void kernel_launch(void* const* d_in, const int* in_sizes, int n_in,
                              void* d_out, int out_size, void* d_ws, size_t ws_size,
                              hipStream_t stream) {
    (void)in_sizes; (void)n_in; (void)out_size; (void)ws_size;
    const float* x   = (const float*)d_in[0];
    const float* w3  = (const float*)d_in[1];
    const float* w1  = (const float*)d_in[2];
    const float* b11 = (const float*)d_in[3];
    const float* b12 = (const float*)d_in[4];
    const float* b13 = (const float*)d_in[5];
    const float* b21 = (const float*)d_in[6];
    const float* b22 = (const float*)d_in[7];
    const float* b23 = (const float*)d_in[8];
    const float* g1  = (const float*)d_in[9];
    const float* be1 = (const float*)d_in[10];
    const float* m1  = (const float*)d_in[11];
    const float* v1  = (const float*)d_in[12];
    const float* g2  = (const float*)d_in[13];
    const float* be2 = (const float*)d_in[14];
    const float* m2  = (const float*)d_in[15];
    const float* v2  = (const float*)d_in[16];
    const float* a1  = (const float*)d_in[17];
    const float* a2  = (const float*)d_in[18];

    uint8_t* ws = (uint8_t*)d_ws;
    const size_t o_a1   = 0;
    const size_t o_obin = o_a1   + (size_t)NB * HP * WP * 256;
    const size_t o_o1f  = o_obin + (size_t)NB * HH * WW * 256;
    const size_t o_B3   = o_o1f  + (size_t)NB * CCH * HH * WW * 4;
    const size_t o_B1   = o_B3   + 589824;
    const size_t o_c    = o_B1   + 65536;

    uint32_t* a1bin = (uint32_t*)(ws + o_a1);
    int8_t*   obin  = (int8_t*)  (ws + o_obin);
    float*    o1f   = (float*)   (ws + o_o1f);
    uint32_t* B3    = (uint32_t*)(ws + o_B3);
    uint32_t* B1    = (uint32_t*)(ws + o_B1);
    float*    sc1   = (float*)(ws + o_c);
    float*    bs1   = sc1 + 256;
    float*    sc2   = bs1 + 256;
    float*    bs2   = sc2 + 256;

    k_consts<<<1, 256, 0, stream>>>(w3, w1, g1, be1, m1, v1, g2, be2, m2, v2,
                                    sc1, bs1, sc2, bs2);
    k_pack3<<<576, 256, 0, stream>>>(w3, B3);
    k_pack1<<<64, 256, 0, stream>>>(w1, B1);
    k_binact<<<(NB * HP * WP * 64) / 256, 256, 0, stream>>>(x, b11, a1bin);

    dim3 gconv(WW / 8, HH / 4, NB);
    k_conv1<<<gconv, 256, 0, stream>>>((const uint8_t*)a1bin, (const int*)B3, x,
                                       sc1, bs1, b12, b13, a1, b21, o1f, obin);
    k_conv2<<<gconv, 256, 0, stream>>>((const uint8_t*)obin, (const int*)B1, o1f,
                                       sc2, bs2, b22, b23, a2, (float*)d_out);
}
